// MultiHeadAttentionBlock_12850542150387
// MI455X (gfx1250) — compile-verified, matrix operand order checked by reading, not run on hardware
//
#include <hip/hip_runtime.h>
#include <hip/hip_bf16.h>

// ---------------------------------------------------------------------------
// MHA block for MI455X (gfx1250, wave32, WMMA).
//   B=2, S=4096, D_MODEL=512, H=8, DK=64.
// All GEMMs run as v_wmma_f32_16x16x32_bf16 (fp32 accumulate); attention is
// flash-style so the [B,H,S,S] score tensor never touches HBM.
// GEMM weights are staged in LDS (one 64KB tile per workgroup); each wave
// owns 32 output rows so every B fragment feeds two WMMAs.
// ---------------------------------------------------------------------------

typedef __bf16 v16bf __attribute__((ext_vector_type(16)));
typedef float  v8f   __attribute__((ext_vector_type(8)));

#define D_MODEL 512
#define NH      8
#define DK      64
#define SEQ     4096
#define BATCH   2
#define MROWS   (BATCH * SEQ)   // 8192

__device__ __forceinline__ __bf16 f32_to_bf16(float f) {
  // round-to-nearest-even bf16
  unsigned u = __builtin_bit_cast(unsigned, f);
  unsigned r = u + 0x7FFFu + ((u >> 16) & 1u);
  unsigned short h = (unsigned short)(r >> 16);
  return __builtin_bit_cast(__bf16, h);
}

// 16-bit WMMA A/B fragment (16x32) loader from a row-major bf16 matrix
// (global or LDS). Layout (contiguous k order, used for A and B alike): lane L<16 holds row (row0+L),
// K = k0..k0+15 in the 16 vector elements; lanes 16..31 hold K = k0+16..k0+31.
// One contiguous, 32B-aligned 32-byte read per lane.
__device__ __forceinline__ v16bf load_frag16(const __bf16* base,
                                             int ld, int row0, int k0,
                                             int l16, int lh) {
  const __bf16* p = base + (size_t)(row0 + l16) * (size_t)ld + (size_t)(k0 + lh * 16);
  return *(const v16bf*)p;
}

__device__ __forceinline__ v8f wmma_bf16(v16bf a, v16bf b, v8f c) {
  return __builtin_amdgcn_wmma_f32_16x16x32_bf16(
      /*neg_a=*/false, a, /*neg_b=*/false, b,
      /*c_mod=*/(short)0, c, /*reuse_a=*/false, /*reuse_b=*/false);
}

// ---------------------------------------------------------------------------
// Kernel 0: fp32 -> bf16 elementwise conversion into workspace.
// ---------------------------------------------------------------------------
__global__ __launch_bounds__(256) void cvt_f32_bf16_kernel(
    const float* __restrict__ in, __bf16* __restrict__ out, int n) {
  int i = blockIdx.x * blockDim.x + threadIdx.x;
  if (i < n) out[i] = f32_to_bf16(in[i]);
}

// ---------------------------------------------------------------------------
// Kernel 1: projection GEMM  Y[M,512] = X[M,512] @ W[512,512]^T + bias.
// 256 threads = 8 waves. Workgroup tile: 256 rows x 64 cols.
// The 64x512 weight tile (64 KB) is staged in LDS once, then every wave
// computes a 32x64 output tile: per k-step 2 A frags (global) x 4 B frags
// (LDS) -> 8 WMMAs.
// mode 0: bf16 out, head-split [B,H,S,DK]             (Q, K projections)
// mode 1: bf16 out, head-split transposed [B,H,DK,S]  (V projection)
// mode 2: f32 out, row-major [M,512]                  (output proj -> d_out)
// ---------------------------------------------------------------------------
__global__ __launch_bounds__(256) void proj_gemm_kernel(
    const __bf16* __restrict__ X, const __bf16* __restrict__ W,
    const float* __restrict__ bias, void* __restrict__ out, int mode) {
  __shared__ __bf16 wsh[64 * D_MODEL];   // 64 KB weight tile, row-major ld=512

  const int tid  = threadIdx.x;
  const int lane = tid & 31;
  const int wave = tid >> 5;
  const int l16 = lane & 15, lh = lane >> 4;

  const int tn = blockIdx.x & 7;   // 8 column tiles of 64
  const int tm = blockIdx.x >> 3;  // 32 row tiles of 256
  const int row0 = tm * 256 + wave * 32;
  const int col0 = tn * 64;

  // ---- cooperative stage of W[col0..col0+63][0..511] into LDS ----
  // 64 rows x 64 chunks of 16B = 4096 chunks; 256 threads x 16 iters.
#pragma unroll
  for (int i = 0; i < 16; ++i) {
    const int t = tid + i * 256;          // 0..4095
    const int r = t >> 6;                 // weight row in tile (0..63)
    const int c = (t & 63) << 3;          // element offset (0..511, step 8)
    *(uint4*)(wsh + ((size_t)r << 9) + c) =
        *(const uint4*)(W + (size_t)(col0 + r) * D_MODEL + c);
  }
  __syncthreads();

  v8f acc[8] = {{}, {}, {}, {}, {}, {}, {}, {}};

#pragma unroll 2
  for (int k0 = 0; k0 < D_MODEL; k0 += 32) {
    // prefetch next A k-tile (global_prefetch_b8; speculative, OOB harmless)
    __builtin_prefetch(X + (size_t)(row0 + l16) * D_MODEL + k0 + 32, 0, 3);
    v16bf a0 = load_frag16(X, D_MODEL, row0,      k0, l16, lh);
    v16bf a1 = load_frag16(X, D_MODEL, row0 + 16, k0, l16, lh);
    v16bf b0 = load_frag16(wsh, D_MODEL,  0, k0, l16, lh);   // ds_load
    v16bf b1 = load_frag16(wsh, D_MODEL, 16, k0, l16, lh);
    v16bf b2 = load_frag16(wsh, D_MODEL, 32, k0, l16, lh);
    v16bf b3 = load_frag16(wsh, D_MODEL, 48, k0, l16, lh);
    acc[0] = wmma_bf16(a0, b0, acc[0]);
    acc[1] = wmma_bf16(a0, b1, acc[1]);
    acc[2] = wmma_bf16(a0, b2, acc[2]);
    acc[3] = wmma_bf16(a0, b3, acc[3]);
    acc[4] = wmma_bf16(a1, b0, acc[4]);
    acc[5] = wmma_bf16(a1, b1, acc[5]);
    acc[6] = wmma_bf16(a1, b2, acc[6]);
    acc[7] = wmma_bf16(a1, b3, acc[7]);
  }

  // Epilogue. C layout: lane L holds column l16, rows r + 8*lh.
#pragma unroll
  for (int t = 0; t < 2; ++t) {
#pragma unroll
    for (int g = 0; g < 4; ++g) {
      const int col = col0 + g * 16 + l16;
      const float bv = bias[col];
#pragma unroll
      for (int r = 0; r < 8; ++r) {
        const int row = row0 + t * 16 + r + 8 * lh;
        const float val = acc[t * 4 + g][r] + bv;
        if (mode == 2) {
          ((float*)out)[(size_t)row * D_MODEL + col] = val;
        } else {
          const int b = row >> 12;            // row / SEQ
          const int s = row & (SEQ - 1);
          const int h = col >> 6;             // col / DK
          const int d = col & (DK - 1);
          __bf16* ob = (__bf16*)out;
          if (mode == 0)
            ob[((size_t)(b * NH + h) * SEQ + s) * DK + d] = f32_to_bf16(val);
          else  // V transposed: [B,H,DK,S] so PV B-fragments load contiguously
            ob[((size_t)(b * NH + h) * DK + d) * SEQ + s] = f32_to_bf16(val);
        }
      }
    }
  }
}

// ---------------------------------------------------------------------------
// Online-softmax update for one 16-row tile against a 32-key block.
// s0/s1: score accumulators (cols l16 / l16+16 of the block, rows r+8*lh).
// Updates running max m[8], sum l[8], rescales the 4 ctx accumulators, and
// stages exp(P) as bf16 into this wave's LDS region (row-major 16x32).
// ---------------------------------------------------------------------------
__device__ __forceinline__ void softmax_update(
    const v8f& s0, const v8f& s1, float* m, float* l, v8f* acc4,
    __bf16* ldsP, int l16, int lh, float scale) {
#pragma unroll
  for (int r = 0; r < 8; ++r) {
    float a0 = s0[r] * scale;
    float a1 = s1[r] * scale;
    float bm = fmaxf(a0, a1);
#pragma unroll
    for (int off = 8; off; off >>= 1)          // row max across 16 lanes
      bm = fmaxf(bm, __shfl_xor(bm, off, 16));
    const float mn = fmaxf(m[r], bm);
    const float al = __expf(m[r] - mn);        // rescale factor for old state
    const float p0 = __expf(a0 - mn);
    const float p1 = __expf(a1 - mn);
    float ps = p0 + p1;
#pragma unroll
    for (int off = 8; off; off >>= 1)          // row sum across 16 lanes
      ps += __shfl_xor(ps, off, 16);
    l[r] = l[r] * al + ps;
    m[r] = mn;
    acc4[0][r] *= al; acc4[1][r] *= al; acc4[2][r] *= al; acc4[3][r] *= al;
    ldsP[(r + 8 * lh) * 32 + l16]      = f32_to_bf16(p0);
    ldsP[(r + 8 * lh) * 32 + l16 + 16] = f32_to_bf16(p1);
  }
}

// ---------------------------------------------------------------------------
// Kernel 2: flash attention. One wave = 32 query rows (two 16-row tiles), so
// each K/V fragment load feeds two WMMAs. 8 waves/WG = 256 query rows.
// Keys streamed in blocks of 32: 8 score WMMAs + online softmax + 8 PV WMMAs
// per block per wave. V fragments are issued before the VALU-heavy softmax so
// their latency hides behind v_exp/shuffle work.
// ---------------------------------------------------------------------------
__global__ __launch_bounds__(256) void flash_attn_kernel(
    const __bf16* __restrict__ Qh,   // [B,H,S,DK]
    const __bf16* __restrict__ Kh,   // [B,H,S,DK]
    const __bf16* __restrict__ Vt,   // [B,H,DK,S]
    __bf16* __restrict__ ctx) {      // [B,S,H*DK]
  __shared__ __bf16 plds[8 * 1024];  // 2 KB per wave: two 16x32 bf16 P tiles

  const int lane = threadIdx.x & 31;
  const int wave = threadIdx.x >> 5;
  const int l16 = lane & 15, lh = lane >> 4;

  const int qt = blockIdx.x & 15;    // 16 query tiles of 256 rows
  const int bh = blockIdx.x >> 4;    // 0..15  (b*8 + h)
  const int b = bh >> 3, h = bh & 7;

  const __bf16* Q = Qh + (size_t)bh * SEQ * DK;
  const __bf16* K = Kh + (size_t)bh * SEQ * DK;
  const __bf16* V = Vt + (size_t)bh * DK * SEQ;

  const int qrow0 = qt * 256 + wave * 32;

  // Two query tiles (rows qrow0.. and qrow0+16..), kept in registers.
  const v16bf qa0 = load_frag16(Q, DK, qrow0,      0,  l16, lh);
  const v16bf qa1 = load_frag16(Q, DK, qrow0,      32, l16, lh);
  const v16bf qb0 = load_frag16(Q, DK, qrow0 + 16, 0,  l16, lh);
  const v16bf qb1 = load_frag16(Q, DK, qrow0 + 16, 32, l16, lh);

  v8f acc[8] = {{}, {}, {}, {}, {}, {}, {}, {}};  // ctx: 2 tiles x 16x64 f32
  float mA[8], lA[8], mB[8], lB[8];
#pragma unroll
  for (int r = 0; r < 8; ++r) {
    mA[r] = -1e30f; lA[r] = 0.0f;
    mB[r] = -1e30f; lB[r] = 0.0f;
  }

  __bf16* wlds = plds + wave * 1024;   // [0..511]=tile A P, [512..1023]=tile B
  const float scale = 0.125f;          // 1/sqrt(DK)

  for (int kb = 0; kb < SEQ; kb += 32) {
    // prefetch next key block (speculative)
    __builtin_prefetch(K + (size_t)(kb + 32 + l16) * DK, 0, 3);
    __builtin_prefetch(V + (size_t)l16 * SEQ + kb + 32, 0, 3);

    // ---- K fragments for this 32-key block ----
    v16bf k00 = load_frag16(K, DK, kb,      0,  l16, lh);
    v16bf k01 = load_frag16(K, DK, kb,      32, l16, lh);
    v16bf k10 = load_frag16(K, DK, kb + 16, 0,  l16, lh);
    v16bf k11 = load_frag16(K, DK, kb + 16, 32, l16, lh);

    // ---- V fragments issued early; consumed after softmax ----
    v16bf v0 = load_frag16(V, SEQ, 0,  kb, l16, lh);
    v16bf v1 = load_frag16(V, SEQ, 16, kb, l16, lh);
    v16bf v2 = load_frag16(V, SEQ, 32, kb, l16, lh);
    v16bf v3 = load_frag16(V, SEQ, 48, kb, l16, lh);

    // ---- scores: 2 query tiles x 16x32 keys = 8 WMMAs ----
    v8f sA0 = {}, sA1 = {}, sB0 = {}, sB1 = {};
    sA0 = wmma_bf16(qa0, k00, sA0);
    sA0 = wmma_bf16(qa1, k01, sA0);
    sA1 = wmma_bf16(qa0, k10, sA1);
    sA1 = wmma_bf16(qa1, k11, sA1);
    sB0 = wmma_bf16(qb0, k00, sB0);
    sB0 = wmma_bf16(qb1, k01, sB0);
    sB1 = wmma_bf16(qb0, k10, sB1);
    sB1 = wmma_bf16(qb1, k11, sB1);

    // ---- online softmax (VALU-heavy; overlaps in-flight V loads) ----
    softmax_update(sA0, sA1, mA, lA, acc,     wlds,       l16, lh, scale);
    softmax_update(sB0, sB1, mB, lB, acc + 4, wlds + 512, l16, lh, scale);

    // reload P in A-fragment layout (same-wave DS ordering; hw waits DScnt)
    const v16bf pfA = *(const v16bf*)(wlds +       l16 * 32 + lh * 16);
    const v16bf pfB = *(const v16bf*)(wlds + 512 + l16 * 32 + lh * 16);

    // ---- ctx += P @ V : each V fragment feeds both query tiles ----
    acc[0] = wmma_bf16(pfA, v0, acc[0]);
    acc[4] = wmma_bf16(pfB, v0, acc[4]);
    acc[1] = wmma_bf16(pfA, v1, acc[1]);
    acc[5] = wmma_bf16(pfB, v1, acc[5]);
    acc[2] = wmma_bf16(pfA, v2, acc[2]);
    acc[6] = wmma_bf16(pfB, v2, acc[6]);
    acc[3] = wmma_bf16(pfA, v3, acc[3]);
    acc[7] = wmma_bf16(pfB, v3, acc[7]);
  }

  // normalize and write ctx in [B,S,H*DK] bf16 (input of output projection)
#pragma unroll
  for (int t = 0; t < 2; ++t) {
    const float* lt = (t == 0) ? lA : lB;
#pragma unroll
    for (int g = 0; g < 4; ++g) {
      const int col = h * DK + g * 16 + l16;
#pragma unroll
      for (int r = 0; r < 8; ++r) {
        const int row = qrow0 + t * 16 + r + 8 * lh;
        const float val = acc[t * 4 + g][r] / lt[r];
        ctx[((size_t)(b * SEQ + row)) * D_MODEL + col] = f32_to_bf16(val);
      }
    }
  }
}

// ---------------------------------------------------------------------------
// Host-side orchestration (graph-capture safe: only kernel launches on stream)
// ---------------------------------------------------------------------------
extern "C" void kernel_launch(void* const* d_in, const int* in_sizes, int n_in,
                              void* d_out, int out_size, void* d_ws, size_t ws_size,
                              hipStream_t stream) {
  (void)in_sizes; (void)n_in; (void)out_size; (void)ws_size;

  const float* q   = (const float*)d_in[0];
  const float* k   = (const float*)d_in[1];
  const float* v   = (const float*)d_in[2];
  /* d_in[3] = mask: numerically dead in the reference (masked_fill discarded) */
  const float* w_q = (const float*)d_in[4];
  const float* b_q = (const float*)d_in[5];
  const float* w_k = (const float*)d_in[6];
  const float* b_k = (const float*)d_in[7];
  const float* w_v = (const float*)d_in[8];
  const float* b_v = (const float*)d_in[9];
  const float* w_o = (const float*)d_in[10];
  const float* b_o = (const float*)d_in[11];

  const size_t XN = (size_t)MROWS * D_MODEL;    // 8192*512
  const size_t WN = (size_t)D_MODEL * D_MODEL;  // 512*512

  char* p = (char*)d_ws;
  __bf16* qb  = (__bf16*)p; p += XN * 2;
  __bf16* kb  = (__bf16*)p; p += XN * 2;
  __bf16* vb  = (__bf16*)p; p += XN * 2;
  __bf16* wqb = (__bf16*)p; p += WN * 2;
  __bf16* wkb = (__bf16*)p; p += WN * 2;
  __bf16* wvb = (__bf16*)p; p += WN * 2;
  __bf16* wob = (__bf16*)p; p += WN * 2;
  __bf16* Qh  = (__bf16*)p; p += XN * 2;   // [B,H,S,DK]
  __bf16* Kh  = (__bf16*)p; p += XN * 2;   // [B,H,S,DK]
  __bf16* Vt  = (__bf16*)p; p += XN * 2;   // [B,H,DK,S]
  __bf16* ctx = (__bf16*)p; p += XN * 2;   // [B,S,H*DK]

  const int CT = 256;
  cvt_f32_bf16_kernel<<<(int)((XN + CT - 1) / CT), CT, 0, stream>>>(q,   qb,  (int)XN);
  cvt_f32_bf16_kernel<<<(int)((XN + CT - 1) / CT), CT, 0, stream>>>(k,   kb,  (int)XN);
  cvt_f32_bf16_kernel<<<(int)((XN + CT - 1) / CT), CT, 0, stream>>>(v,   vb,  (int)XN);
  cvt_f32_bf16_kernel<<<(int)((WN + CT - 1) / CT), CT, 0, stream>>>(w_q, wqb, (int)WN);
  cvt_f32_bf16_kernel<<<(int)((WN + CT - 1) / CT), CT, 0, stream>>>(w_k, wkb, (int)WN);
  cvt_f32_bf16_kernel<<<(int)((WN + CT - 1) / CT), CT, 0, stream>>>(w_v, wvb, (int)WN);
  cvt_f32_bf16_kernel<<<(int)((WN + CT - 1) / CT), CT, 0, stream>>>(w_o, wob, (int)WN);

  // (M/256)*(N/64) = 32*8 = 256 workgroups per GEMM, 256 threads each
  proj_gemm_kernel<<<256, 256, 0, stream>>>(qb, wqb, b_q, (void*)Qh, 0);
  proj_gemm_kernel<<<256, 256, 0, stream>>>(kb, wkb, b_k, (void*)Kh, 0);
  proj_gemm_kernel<<<256, 256, 0, stream>>>(vb, wvb, b_v, (void*)Vt, 1);

  // 16 (b,h) * 16 query tiles = 256 workgroups
  flash_attn_kernel<<<256, 256, 0, stream>>>(Qh, Kh, Vt, ctx);

  proj_gemm_kernel<<<256, 256, 0, stream>>>(ctx, wob, b_o, d_out, 2);
}
